// MSAPairWeightedAveraging_24412594111337
// MI455X (gfx1250) — matrix operand order repaired and checked by reading, not compiled, not run on hardware
//
#include <hip/hip_runtime.h>
#include <hip/hip_bf16.h>
#include <math.h>

typedef __bf16 bf16;
typedef bf16 v16bf __attribute__((ext_vector_type(16)));
typedef bf16 v8bf  __attribute__((ext_vector_type(8)));
typedef float v8f  __attribute__((ext_vector_type(8)));

#define DEV __device__ __forceinline__

constexpr int S_  = 512;
constexpr int N_  = 384;
constexpr int CIN = 64;
constexpr int CZ  = 128;
constexpr int H_  = 8;
constexpr int C_  = 32;
constexpr int D_  = 256;   // H*C
constexpr float EPS_ = 1e-5f;
constexpr float INF_ = 1e9f;

DEV float fast_rcp(float x) { return __builtin_amdgcn_rcpf(x); }
DEV float fast_sigmoid(float x) { return fast_rcp(1.f + __expf(-x)); }

// ---- WMMA fragment builders (wave32, 16-bit operand layouts per CDNA5 ISA 7.12.2) ----
DEV v16bf pack16(const bf16* p0, const bf16* p1) {
  v8bf lo = *(const v8bf*)p0;
  v8bf hi = *(const v8bf*)p1;
  return __builtin_shufflevector(lo, hi, 0,1,2,3,4,5,6,7,8,9,10,11,12,13,14,15);
}
// A matrix 16(M)x32(K), row-major source with leading dim lda (elements).
DEV v16bf load_fragA(const bf16* base, int lda, int laneN, int sel) {
  const bf16* p = base + laneN * lda + sel * 8;
  return pack16(p, p + 16);
}
// B matrix 32(K)x16(N), stored N-major: row n holds K contiguous, leading dim ldb.
DEV v16bf load_fragB(const bf16* base, int ldb, int laneN, int sel) {
  const bf16* p = base + laneN * ldb + sel * 8;
  return pack16(p, p + 16);
}

// ================= Kernel 1: pair branch LN + logits + softmax =================
__global__ void __launch_bounds__(256)
pair_softmax_kernel(const float* __restrict__ z, const float* __restrict__ mask,
                    const float* __restrict__ gz, const float* __restrict__ bz_ln,
                    const float* __restrict__ wz, const float* __restrict__ bz,
                    bf16* __restrict__ watt) {
  __shared__ float s_wz[CZ * H_];
  __shared__ float s_g[CZ], s_b[CZ];
  __shared__ float s_logit[H_ * N_];

  const int q = blockIdx.x;
  const int tid = threadIdx.x;

  for (int i = tid; i < CZ * H_; i += 256) s_wz[i] = wz[i];
  for (int i = tid; i < CZ; i += 256) { s_g[i] = gz[i]; s_b[i] = bz_ln[i]; }
  __syncthreads();

  for (int k = tid; k < N_; k += 256) {
    const float* zp = z + ((size_t)q * N_ + k) * CZ;
    float sum = 0.f, sq = 0.f;
    for (int i = 0; i < CZ; ++i) { float x = zp[i]; sum += x; sq += x * x; }
    float mu = sum * (1.f / CZ);
    float var = sq * (1.f / CZ) - mu * mu;
    float rinv = rsqrtf(var + EPS_);
    float acc[H_];
#pragma unroll
    for (int h = 0; h < H_; ++h) acc[h] = 0.f;
    for (int i = 0; i < CZ; ++i) {
      float zn = (zp[i] - mu) * rinv * s_g[i] + s_b[i];
#pragma unroll
      for (int h = 0; h < H_; ++h) acc[h] += zn * s_wz[i * H_ + h];
    }
    float bias = INF_ * (mask[q * N_ + k] - 1.f);
#pragma unroll
    for (int h = 0; h < H_; ++h) s_logit[h * N_ + k] = acc[h] + bz[h] + bias;
  }
  __syncthreads();

  const int lane = tid & 31;
  const int h = __builtin_amdgcn_readfirstlane(tid >> 5);   // 8 waves, one head each
  float e[12];
  float mx = -3.4e38f;
#pragma unroll
  for (int j = 0; j < 12; ++j) { float v = s_logit[h * N_ + lane + j * 32]; e[j] = v; mx = fmaxf(mx, v); }
  for (int off = 16; off; off >>= 1) mx = fmaxf(mx, __shfl_xor(mx, off));
  float sm = 0.f;
#pragma unroll
  for (int j = 0; j < 12; ++j) { e[j] = __expf(e[j] - mx); sm += e[j]; }
  for (int off = 16; off; off >>= 1) sm += __shfl_xor(sm, off);
  float r = fast_rcp(sm);
  bf16* wp = watt + ((size_t)h * N_ + q) * N_;
#pragma unroll
  for (int j = 0; j < 12; ++j) wp[lane + j * 32] = (bf16)(e[j] * r);
}

// ====== Kernel 2: weight prep: w_o^T bf16 [64][256], [w_v;w_g]^T bf16 [512][64] ======
__global__ void prep_weights_kernel(const float* __restrict__ wo, const float* __restrict__ wv,
                                    const float* __restrict__ wg,
                                    bf16* __restrict__ wot, bf16* __restrict__ wcat) {
  int i = blockIdx.x * 256 + threadIdx.x;
  if (i < CIN * D_) {            // wot[j][k] = wo[k][j]
    int j = i / D_, k = i % D_;
    wot[i] = (bf16)wo[k * CIN + j];
  }
  if (i < 2 * D_ * CIN) {        // wcat[r][c]: r<256 -> w_v[:,r]^T ; r>=256 -> w_g[:,r-256]^T
    int r = i / CIN, c = i % CIN;
    wcat[i] = (r < D_) ? (bf16)wv[c * D_ + r] : (bf16)wg[c * D_ + (r - D_)];
  }
}

// ====== Kernel 3: LN(m) + fused value/gate projections via WMMA ======
// grid = S*N/16 blocks; block handles a 16-row tile of mn, 8 waves x 4 n-tiles (32 n-tiles)
__global__ void __launch_bounds__(256)
msa_proj_kernel(const float* __restrict__ m, const float* __restrict__ gm,
                const float* __restrict__ bm, const bf16* __restrict__ wcat,
                const float* __restrict__ bv, const float* __restrict__ bg,
                bf16* __restrict__ vws, bf16* __restrict__ g_ws) {
  __shared__ float s_x[16 * CIN];
  __shared__ float s_mu[16], s_ri[16];
  __shared__ __align__(16) bf16 s_a[16 * CIN];   // A fragment source, lda = 64

  const int tid = threadIdx.x;
  const size_t row0 = (size_t)blockIdx.x * 16;

  for (int i = tid; i < 16 * CIN; i += 256) s_x[i] = m[row0 * CIN + i];
  __syncthreads();

  if (tid < 16) {
    float sum = 0.f, sq = 0.f;
    const float* xr = &s_x[tid * CIN];
    for (int i = 0; i < CIN; ++i) { float x = xr[i]; sum += x; sq += x * x; }
    float mu = sum * (1.f / CIN);
    s_mu[tid] = mu;
    s_ri[tid] = rsqrtf(sq * (1.f / CIN) - mu * mu + EPS_);
  }
  __syncthreads();

  for (int i = tid; i < 16 * CIN; i += 256) {
    int rr = i >> 6, c = i & 63;
    s_a[i] = (bf16)((s_x[i] - s_mu[rr]) * s_ri[rr] * gm[c] + bm[c]);
  }
  __syncthreads();

  const int lane = tid & 31;
  const int wave = __builtin_amdgcn_readfirstlane(tid >> 5);   // SGPR -> scalar branches
  const int laneN = lane & 15, sel = lane >> 4;
  const int s = (int)(row0 / N_);
  const int kbase = (int)(row0 % N_);            // N_ % 16 == 0 -> tile never crosses s

  // A fragments are loop-invariant (LDS)
  v16bf a0 = load_fragA(&s_a[0], CIN, laneN, sel);
  v16bf a1 = load_fragA(&s_a[32], CIN, laneN, sel);

#pragma unroll
  for (int j = 0; j < 4; ++j) {
    const int nt = wave * 4 + j;                 // 0..31; <16 => value, >=16 => gate
    v16bf b0 = load_fragB(wcat + (size_t)nt * 16 * CIN + 0, CIN, laneN, sel);
    v16bf b1 = load_fragB(wcat + (size_t)nt * 16 * CIN + 32, CIN, laneN, sel);
    v8f acc = {};
    acc = __builtin_amdgcn_wmma_f32_16x16x32_bf16(false, a0, false, b0, (short)0, acc, false, false);
    acc = __builtin_amdgcn_wmma_f32_16x16x32_bf16(false, a1, false, b1, (short)0, acc, false, false);
    if (nt < 16) {
      const int d = nt * 16 + laneN;             // output col in [0,256)
      const int hh = d >> 5, c = d & 31;
      const float bias = bv[d];
      bf16* vp = vws + (((size_t)s * H_ + hh) * C_ + c) * N_ + kbase;
#pragma unroll
      for (int rr = 0; rr < 8; ++rr) vp[rr + sel * 8] = (bf16)(acc[rr] + bias);
    } else {
      const int d = (nt - 16) * 16 + laneN;
      const float bias = bg[d];
      bf16* gp = g_ws + row0 * D_ + d;
#pragma unroll
      for (int rr = 0; rr < 8; ++rr)
        gp[(size_t)(rr + sel * 8) * D_] = (bf16)fast_sigmoid(acc[rr] + bias);
    }
  }
}

// ==== Kernel 4: attention einsum (WMMA) + precomputed gate + output projection (WMMA) ====
__global__ void __launch_bounds__(256)
attn_out_kernel(const bf16* __restrict__ watt, const bf16* __restrict__ vws,
                const bf16* __restrict__ g_ws, const bf16* __restrict__ wot,
                const float* __restrict__ bo, float* __restrict__ out) {
  __shared__ __align__(16) bf16 s_gA[16 * D_];   // gated output, A operand of final GEMM
  __shared__ __align__(16) bf16 s_wo[CIN * D_];  // w_o^T staged via async-to-LDS (32 KB)

  const int qt = blockIdx.x, s = blockIdx.y;
  const int q0 = qt * 16;
  const int tid = threadIdx.x, lane = tid & 31;
  const int wave = __builtin_amdgcn_readfirstlane(tid >> 5);
  const int laneN = lane & 15, sel = lane >> 4;

  // Stage w_o^T into LDS asynchronously (ASYNCcnt); overlaps the main WMMA loop.
  for (int c = tid; c < (CIN * D_) / 8; c += 256) {
    unsigned ldsoff = (unsigned)(uintptr_t)&s_wo[c * 8];
    const bf16* gp = wot + c * 8;
    asm volatile("global_load_async_to_lds_b128 %0, %1, off"
                 :: "v"(ldsoff), "v"(gp) : "memory");
  }

  v8f acc[2] = {};
  const int h = wave;
  const bf16* Wp = watt + ((size_t)h * N_ + q0) * N_;       // A: [16q x 384k], lda=N
  const bf16* Vp = vws + ((size_t)s * H_ + h) * (C_ * N_);  // B: [32c x 384k] N-major

  for (int k0 = 0; k0 < N_; k0 += 32) {
    if (k0 + 32 < N_) {
      __builtin_prefetch(Wp + laneN * N_ + k0 + 32, 0, 3);
      __builtin_prefetch(Vp + laneN * N_ + k0 + 32, 0, 3);
    }
    v16bf a  = load_fragA(Wp + k0, N_, laneN, sel);
    v16bf b0 = load_fragB(Vp + k0, N_, laneN, sel);
    v16bf b1 = load_fragB(Vp + 16 * N_ + k0, N_, laneN, sel);
    acc[0] = __builtin_amdgcn_wmma_f32_16x16x32_bf16(false, a, false, b0, (short)0, acc[0], false, false);
    acc[1] = __builtin_amdgcn_wmma_f32_16x16x32_bf16(false, a, false, b1, (short)0, acc[1], false, false);
  }

  // apply precomputed sigmoid gate (bf16 loads), pack gated output into LDS
  const bf16* grow = g_ws + ((size_t)s * N_ + q0) * D_;
#pragma unroll
  for (int ch = 0; ch < 2; ++ch) {
    const int d = h * C_ + ch * 16 + laneN;
#pragma unroll
    for (int rr = 0; rr < 8; ++rr) {
      const int mrow = rr + sel * 8;
      float gate = (float)grow[(size_t)mrow * D_ + d];
      s_gA[mrow * D_ + d] = (bf16)(acc[ch][rr] * gate);
    }
  }

  asm volatile("s_wait_asynccnt 0x0" ::: "memory");
  __syncthreads();

  // final projection: [16q x 256d] @ [256d x 64c_out]; waves 0..3 own 16-wide n-tiles
  if (wave < 4) {
    v8f o = {};
    const bf16* Bp = &s_wo[wave * 16 * D_];
#pragma unroll
    for (int k0 = 0; k0 < D_; k0 += 32) {
      v16bf a = load_fragA(&s_gA[k0], D_, laneN, sel);
      v16bf b = load_fragB(Bp + k0, D_, laneN, sel);
      o = __builtin_amdgcn_wmma_f32_16x16x32_bf16(false, a, false, b, (short)0, o, false, false);
    }
    const int j = wave * 16 + laneN;
    const float bov = bo[j];
#pragma unroll
    for (int rr = 0; rr < 8; ++rr) {
      int mrow = rr + sel * 8;
      out[((size_t)s * N_ + q0 + mrow) * CIN + j] = o[rr] + bov;
    }
  }
}

extern "C" void kernel_launch(void* const* d_in, const int* in_sizes, int n_in,
                              void* d_out, int out_size, void* d_ws, size_t ws_size,
                              hipStream_t stream) {
  (void)in_sizes; (void)n_in; (void)out_size; (void)ws_size;
  const float* m     = (const float*)d_in[0];
  const float* z     = (const float*)d_in[1];
  const float* mask  = (const float*)d_in[2];
  const float* gm    = (const float*)d_in[3];
  const float* bm    = (const float*)d_in[4];
  const float* gz    = (const float*)d_in[5];
  const float* bz_ln = (const float*)d_in[6];
  const float* wz    = (const float*)d_in[7];
  const float* bz    = (const float*)d_in[8];
  const float* wv    = (const float*)d_in[9];
  const float* bv    = (const float*)d_in[10];
  const float* wg    = (const float*)d_in[11];
  const float* bg    = (const float*)d_in[12];
  const float* wo    = (const float*)d_in[13];
  const float* bo    = (const float*)d_in[14];
  float* out = (float*)d_out;

  char* ws = (char*)d_ws;
  bf16* watt = (bf16*)ws; ws += (size_t)H_ * N_ * N_ * sizeof(bf16);        // 2.36 MB
  bf16* vws  = (bf16*)ws; ws += (size_t)S_ * H_ * C_ * N_ * sizeof(bf16);   // 100.7 MB
  bf16* g_ws = (bf16*)ws; ws += (size_t)S_ * N_ * D_ * sizeof(bf16);        // 100.7 MB
  bf16* wot  = (bf16*)ws; ws += (size_t)CIN * D_ * sizeof(bf16);            // 32 KB
  bf16* wcat = (bf16*)ws; ws += (size_t)2 * D_ * CIN * sizeof(bf16);        // 64 KB

  prep_weights_kernel<<<(2 * D_ * CIN + 255) / 256, 256, 0, stream>>>(wo, wv, wg, wot, wcat);
  pair_softmax_kernel<<<N_, 256, 0, stream>>>(z, mask, gz, bz_ln, wz, bz, watt);
  msa_proj_kernel<<<(S_ * N_) / 16, 256, 0, stream>>>(m, gm, bm, wcat, bv, bg, vws, g_ws);
  attn_out_kernel<<<dim3(N_ / 16, S_), 256, 0, stream>>>(watt, vws, g_ws, wot, bo, out);
}
